// NGCFConv_5153960755314
// MI455X (gfx1250) — matrix operand order repaired and checked by reading, not compiled, not run on hardware
//
#include <hip/hip_runtime.h>
#include <hip/hip_bf16.h>

#define D 64
#define ALPHA 0.2f
#define L2EPS 1e-12f

typedef __attribute__((ext_vector_type(16))) __bf16 v16bf;
typedef __attribute__((ext_vector_type(8)))  float  v8f;

__device__ __forceinline__ __bf16 to_bf16(float f) {
  return static_cast<__bf16>(f);   // backend: native cvt if available, else inline RNE
}

// ---------------------------------------------------------------- zero scratch
__global__ void zero_kernel(float* __restrict__ p, long long n) {
  long long i = (long long)blockIdx.x * blockDim.x + threadIdx.x;
  if (i < n) p[i] = 0.0f;
}

// ---------------------------------------------------------------- deg[row] += 1
__global__ void deg_kernel(const int* __restrict__ row, int E, float* __restrict__ deg) {
  int e = blockIdx.x * blockDim.x + threadIdx.x;
  if (e < E) atomicAdd(&deg[row[e]], 1.0f);
}

// ---------------------------------------------------------------- deg^-1/2
__global__ void invsqrt_kernel(const float* __restrict__ deg, float* __restrict__ isd, int n) {
  int i = blockIdx.x * blockDim.x + threadIdx.x;
  if (i < n) {
    float d = deg[i];
    isd[i] = (d > 0.0f) ? rsqrtf(fmaxf(d, 1.0f)) : 0.0f;
  }
}

// ---------------------------------------------------------------- pre-pack weights
// Convert W_gcn / W_int to bf16 in WMMA B-fragment order:
//   flat = (((w*2 + kc)*4 + nc)*32 + lane)*16 + e
//   B (32x16 bf16) lane layout: N = nc*16 + (lane&15), K = kc*32 + (lane>=16?8:0) + e + (e>=8?8:0)
__global__ void pack_weights_kernel(const float* __restrict__ Wg,
                                    const float* __restrict__ Wi,
                                    __bf16* __restrict__ pk) {
  int idx = blockIdx.x * blockDim.x + threadIdx.x;   // 0..8191
  if (idx >= 2 * D * D) return;
  int e    = idx & 15;
  int lane = (idx >> 4) & 31;
  int nc   = (idx >> 9) & 3;
  int kc   = (idx >> 11) & 1;
  int w    = (idx >> 12) & 1;
  int K = kc * 32 + ((lane >= 16) ? 8 : 0) + e + ((e >= 8) ? 8 : 0);
  int n = nc * 16 + (lane & 15);
  const float* W = w ? Wi : Wg;
  pk[idx] = to_bf16(W[K * D + n]);
}

// ------------------------------------------------- h[row] += w * x[col]  (16 lanes/edge)
__global__ void scatter_kernel(const float* __restrict__ x,
                               const int* __restrict__ row,
                               const int* __restrict__ col,
                               const float* __restrict__ isd,
                               float* __restrict__ h, int E) {
  long long gid = (long long)blockIdx.x * blockDim.x + threadIdx.x;
  int e = (int)(gid >> 4);
  int t = (int)(gid & 15);
  if (e >= E) return;
  int r = row[e], c = col[e];
  float w = isd[r] * isd[c];
  float4 v = reinterpret_cast<const float4*>(x + (long long)c * D)[t];   // 256B-aligned gather
  float* hp = h + (long long)r * D + t * 4;
  atomicAdd(hp + 0, w * v.x);
  atomicAdd(hp + 1, w * v.y);
  atomicAdd(hp + 2, w * v.z);
  atomicAdd(hp + 3, w * v.w);
}

// ------------------------------------------------- fused dual-GEMM + leaky + l2norm
// One wave per 16-row tile. bf16 WMMA 16x16x32, f32 accumulate.
//
// A (16x32 bf16) lane layout (ISA 7.12.2): lane l, element e ->
//   M = l&15,  K = kc*32 + (l>=16 ? 8:0) + e + (e>=8 ? 8:0)
//   => per lane, two contiguous 8-float runs: [kbase, kbase+8) and [kbase+16, kbase+24)
// C/D (16x16 f32): VGPR r, lane l -> M = r + (l>=16 ? 8:0), N = l&15
__global__ __launch_bounds__(256) void fused_kernel(
    const float* __restrict__ x, const float* __restrict__ h,
    const __bf16* __restrict__ pkW,
    const float* __restrict__ bg, const float* __restrict__ bi,
    float* __restrict__ out, int nTiles, int N) {
  __shared__ float smem[8][16 * 65 + 16];   // per-wave: padded 16x64 tile + 16 scales

  const int lane = threadIdx.x & 31;
  const int wave = threadIdx.x >> 5;
  const int tile = blockIdx.x * 8 + wave;
  const bool active = tile < nTiles;
  const int t2 = active ? tile : 0;         // inactive waves compute garbage, store nothing

  const int laneHiA = (lane >= 16) ? 8 : 0;
  int rowIdx = t2 * 16 + (lane & 15);
  if (rowIdx >= N) rowIdx = N - 1;
  const long long rowBase = (long long)rowIdx * D;

  // A fragments for both branches: a1 = h, a2 = x*h  (vectorized float4 loads)
  v16bf a1[2], a2[2];
#pragma unroll
  for (int kc = 0; kc < 2; ++kc) {
    const int kbase = kc * 32 + laneHiA;
    const float4* hp = reinterpret_cast<const float4*>(h + rowBase + kbase);
    const float4* xp = reinterpret_cast<const float4*>(x + rowBase + kbase);
    float4 hv[4], xv[4];
    hv[0] = hp[0]; hv[1] = hp[1]; hv[2] = hp[4]; hv[3] = hp[5];   // K: +0..7, +16..23
    xv[0] = xp[0]; xv[1] = xp[1]; xv[2] = xp[4]; xv[3] = xp[5];
#pragma unroll
    for (int q = 0; q < 4; ++q) {
      const float* hf = reinterpret_cast<const float*>(&hv[q]);
      const float* xf = reinterpret_cast<const float*>(&xv[q]);
#pragma unroll
      for (int j = 0; j < 4; ++j) {
        int e = q * 4 + j;
        a1[kc][e] = to_bf16(hf[j]);
        a2[kc][e] = to_bf16(xf[j] * hf[j]);
      }
    }
  }

  const int nlo = lane & 15;
  const v16bf* pkv = reinterpret_cast<const v16bf*>(pkW);   // fragment-granular view

#pragma unroll
  for (int nc = 0; nc < 4; ++nc) {
    v8f accg = {0.f, 0.f, 0.f, 0.f, 0.f, 0.f, 0.f, 0.f};
    v8f acci = {0.f, 0.f, 0.f, 0.f, 0.f, 0.f, 0.f, 0.f};
    const int n = nc * 16 + nlo;
#pragma unroll
    for (int kc = 0; kc < 2; ++kc) {
      v16bf bgf = pkv[((0 * 2 + kc) * 4 + nc) * 32 + lane];   // 32B contiguous load
      v16bf bif = pkv[((1 * 2 + kc) * 4 + nc) * 32 + lane];
      accg = __builtin_amdgcn_wmma_f32_16x16x32_bf16(false, a1[kc], false, bgf,
                                                     (short)0, accg, false, false);
      acci = __builtin_amdgcn_wmma_f32_16x16x32_bf16(false, a2[kc], false, bif,
                                                     (short)0, acci, false, false);
    }
    const float bgb = bg[n];
    const float bib = bi[n];
#pragma unroll
    for (int r = 0; r < 8; ++r) {
      float g = accg[r] + bgb; g = (g > 0.f) ? g : ALPHA * g;
      float q = acci[r] + bib; q = (q > 0.f) ? q : ALPHA * q;
      int mrow = r + ((lane >= 16) ? 8 : 0);
      smem[wave][mrow * 65 + n] = g + q;
    }
  }

  __syncthreads();

  // per-row sum of squares -> scale (lanes 0..15, one row each; padded stride 65)
  if (lane < 16) {
    float s = 0.f;
#pragma unroll
    for (int k = 0; k < D; ++k) {
      float v = smem[wave][lane * 65 + k];
      s += v * v;
    }
    smem[wave][16 * 65 + lane] = rsqrtf(fmaxf(s, L2EPS));
  }

  __syncthreads();

  if (active) {
    const long long outBase = (long long)tile * 16 * D;
#pragma unroll
    for (int i = 0; i < 32; ++i) {
      int idx = lane + i * 32;            // 0..1023 coalesced
      int mr = idx >> 6;
      int nc2 = idx & 63;
      int grow = tile * 16 + mr;
      if (grow < N)
        out[outBase + idx] = smem[wave][mr * 65 + nc2] * smem[wave][16 * 65 + mr];
    }
  }
}

// ---------------------------------------------------------------- launcher
extern "C" void kernel_launch(void* const* d_in, const int* in_sizes, int n_in,
                              void* d_out, int out_size, void* d_ws, size_t ws_size,
                              hipStream_t stream) {
  const float* x  = (const float*)d_in[0];
  const int*   ei = (const int*)d_in[1];
  const float* Wg = (const float*)d_in[2];
  const float* bg = (const float*)d_in[3];
  const float* Wi = (const float*)d_in[4];
  const float* bi = (const float*)d_in[5];
  float* out = (float*)d_out;

  const int N = in_sizes[0] / D;
  const int E = in_sizes[1] / 2;
  const int* row = ei;
  const int* col = ei + E;

  // workspace layout: h[N*D] f32 | deg[N] f32 | isd[N] f32 | pkW[2*D*D] bf16 (32B aligned)
  float* h   = (float*)d_ws;
  float* deg = h + (size_t)N * D;
  float* isd = deg + N;
  size_t off = (size_t)((char*)(isd + N) - (char*)d_ws);
  off = (off + 31) & ~(size_t)31;
  __bf16* pkW = (__bf16*)((char*)d_ws + off);

  long long zc = (long long)N * D + N;   // zero h and deg (contiguous)
  zero_kernel<<<(int)((zc + 255) / 256), 256, 0, stream>>>(h, zc);

  pack_weights_kernel<<<(2 * D * D + 255) / 256, 256, 0, stream>>>(Wg, Wi, pkW);

  deg_kernel<<<(E + 255) / 256, 256, 0, stream>>>(row, E, deg);
  invsqrt_kernel<<<(N + 255) / 256, 256, 0, stream>>>(deg, isd, N);

  long long sthreads = (long long)E * 16;
  scatter_kernel<<<(int)((sthreads + 255) / 256), 256, 0, stream>>>(x, row, col, isd, h, E);

  const int nTiles = (N + 15) / 16;
  const int blocks = (nTiles + 7) / 8;
  fused_kernel<<<blocks, 256, 0, stream>>>(x, h, pkW, bg, bi, out, nTiles, N);
}
